// GraphSAGE_22832046146009
// MI455X (gfx1250) — matrix operand order repaired and checked by reading, not compiled, not run on hardware
//
#include <hip/hip_runtime.h>
#include <hip/hip_bf16.h>

// ---------------------------------------------------------------------------
// GraphSAGE (2x SAGEConv 'mean') for gfx1250 / MI455X.
// N = 50000 nodes, E = 600000 edges, D = 128.
//
// Pipeline per layer:
//   deg[dst] += 1                      (once, layer-shared)
//   agg[dst] += feat_bf16[src]        (f32 atomics, gather in bf16)
//   mean_bf16 = bf16(agg / max(deg,1))
//   h = relu?( x @ Wself + mean @ Wneigh + b )  -- WMMA bf16, f32 accum
// ---------------------------------------------------------------------------

#define N_NODES 50000
#define N_EDGES 600000
#define D       128

typedef __attribute__((ext_vector_type(16))) __bf16 v16bf;
typedef __attribute__((ext_vector_type(8)))  __bf16 v8bf;
typedef __attribute__((ext_vector_type(4)))  __bf16 v4bf;
typedef __attribute__((ext_vector_type(8)))  float  v8f;

// --- small prep kernels -----------------------------------------------------

// Convert f32 -> bf16 (elementwise)
__global__ void cast_bf16_kernel(const float* __restrict__ x,
                                 __bf16* __restrict__ y, int n) {
    int i = blockIdx.x * blockDim.x + threadIdx.x;
    if (i < n) y[i] = (__bf16)x[i];
}

// Transpose W[k][n] (row-major DxD f32) into Wt[n][k] bf16 so B-fragments are
// contiguous 16B loads per lane.
__global__ void prep_w_kernel(const float* __restrict__ w,
                              __bf16* __restrict__ wt) {
    int i = blockIdx.x * blockDim.x + threadIdx.x;
    if (i < D * D) {
        int k = i >> 7;      // i / 128
        int n = i & (D - 1); // i % 128
        wt[n * D + k] = (__bf16)w[i];
    }
}

// deg[dst] += 1
__global__ void deg_kernel(const int* __restrict__ dst,
                           float* __restrict__ deg, int nE) {
    int e = blockIdx.x * blockDim.x + threadIdx.x;
    if (e < nE) atomicAdd(&deg[dst[e]], 1.0f);
}

// agg[dst] += feat[src]   (32 threads per edge, 4 channels each)
__global__ __launch_bounds__(256)
void scatter_kernel(const __bf16* __restrict__ feat,
                    const int* __restrict__ src,
                    const int* __restrict__ dst,
                    float* __restrict__ agg, int nE) {
    long long tid = (long long)blockIdx.x * blockDim.x + threadIdx.x;
    int e = (int)(tid >> 5);
    if (e >= nE) return;
    int d0 = ((int)tid & 31) * 4;
    const v4bf v = *(const v4bf*)(feat + (long long)src[e] * D + d0);
    float* ap = agg + (long long)dst[e] * D + d0;
    atomicAdd(ap + 0, (float)v[0]);
    atomicAdd(ap + 1, (float)v[1]);
    atomicAdd(ap + 2, (float)v[2]);
    atomicAdd(ap + 3, (float)v[3]);
}

// mean_bf16 = bf16(agg / max(deg,1))
__global__ void mean_kernel(const float* __restrict__ agg,
                            const float* __restrict__ deg,
                            __bf16* __restrict__ meanbf, int n) {
    int i = blockIdx.x * blockDim.x + threadIdx.x;
    if (i < n) {
        float d = deg[i >> 7];
        meanbf[i] = (__bf16)(agg[i] / fmaxf(d, 1.0f));
    }
}

// --- WMMA GEMM --------------------------------------------------------------
// 16-bit A 16x32 layout (per lane L: m=L%16, h=L/16):
//   v16bf elems 0..7  = A[m][kb + 8h + 0..7]
//   v16bf elems 8..15 = A[m][kb + 16 + 8h + 0..7]
// 16-bit B 32x16 layout (per lane L: n=L%16, h=L/16), with Wt[n][k]:
//   v16bf elems 0..7 = Wt[n][kb + 8h + 0..7], elems 8..15 = Wt[n][kb + 16 + 8h + 0..7]
__device__ __forceinline__ v16bf load_frag(const __bf16* p0, const __bf16* p1) {
    v8bf lo = *(const v8bf*)p0;
    v8bf hi = *(const v8bf*)p1;
    v16bf r;
#pragma unroll
    for (int i = 0; i < 8; ++i) { r[i] = lo[i]; r[8 + i] = hi[i]; }
    return r;
}

template <bool RELU, bool OUT_BF16>
__global__ __launch_bounds__(256)
void sage_gemm_kernel(const __bf16* __restrict__ xbf,
                      const __bf16* __restrict__ meanbf,
                      const __bf16* __restrict__ wt_self,
                      const __bf16* __restrict__ wt_neigh,
                      const float* __restrict__ bias,
                      void* __restrict__ out) {
    const int m0   = blockIdx.x * 16;        // 3125 M-tiles, exact
    const int wave = threadIdx.x >> 5;       // 8 waves -> 8 N-tiles
    const int lane = threadIdx.x & 31;
    const int n0   = wave * 16;
    const int ml   = lane & 15;              // row (A) / col (B,C) index
    const int h    = lane >> 4;

    const __bf16* xrow = xbf    + (long long)(m0 + ml) * D;
    const __bf16* mrow = meanbf + (long long)(m0 + ml) * D;
    const __bf16* wsr  = wt_self  + (n0 + ml) * D;
    const __bf16* wnr  = wt_neigh + (n0 + ml) * D;

    v8f c = {};
#pragma unroll
    for (int kc = 0; kc < 4; ++kc) {
        const int kb = kc * 32;
        v16bf a_s = load_frag(xrow + kb + 8 * h, xrow + kb + 16 + 8 * h);
        v16bf b_s = load_frag(wsr + kb + 8 * h, wsr + kb + 16 + 8 * h);
        c = __builtin_amdgcn_wmma_f32_16x16x32_bf16(false, a_s, false, b_s,
                                                    (short)0, c, false, false);
        v16bf a_m = load_frag(mrow + kb + 8 * h, mrow + kb + 16 + 8 * h);
        v16bf b_n = load_frag(wnr + kb + 8 * h, wnr + kb + 16 + 8 * h);
        c = __builtin_amdgcn_wmma_f32_16x16x32_bf16(false, a_m, false, b_n,
                                                    (short)0, c, false, false);
    }

    const float bv = bias[n0 + ml];

    // All fragment loads in this block are consumed above; barrier makes the
    // in-place (out == xbf) layer-1 store safe across the block's 8 waves.
    __syncthreads();

    // C/D layout: VGPR r, lanes 0-15 -> M=r, lanes 16-31 -> M=r+8; N = lane%16
#pragma unroll
    for (int r = 0; r < 8; ++r) {
        float v = c[r] + bv;
        if (RELU) v = fmaxf(v, 0.0f);
        long long idx = (long long)(m0 + r + 8 * h) * D + n0 + ml;
        if (OUT_BF16) ((__bf16*)out)[idx] = (__bf16)v;
        else          ((float*)out)[idx]  = v;
    }
}

// --- launcher ---------------------------------------------------------------

extern "C" void kernel_launch(void* const* d_in, const int* in_sizes, int n_in,
                              void* d_out, int out_size, void* d_ws, size_t ws_size,
                              hipStream_t stream) {
    const float* in_feat = (const float*)d_in[0];
    const int*   src     = (const int*)d_in[1];
    const int*   dst     = (const int*)d_in[2];
    const float* Ws1     = (const float*)d_in[3];
    const float* Wn1     = (const float*)d_in[4];
    const float* b1      = (const float*)d_in[5];
    const float* Ws2     = (const float*)d_in[6];
    const float* Wn2     = (const float*)d_in[7];
    const float* b2      = (const float*)d_in[8];

    char* ws = (char*)d_ws;
    const size_t WT_SZ   = (size_t)D * D * sizeof(__bf16);      // 32 KB
    const size_t DEG_OFF = 4 * WT_SZ;                           // 131072
    const size_t AGG_OFF = DEG_OFF + 204800;                    // deg padded
    const size_t XBF_OFF = AGG_OFF + (size_t)N_NODES * D * 4;   // agg f32
    const size_t MBF_OFF = XBF_OFF + (size_t)N_NODES * D * 2;   // feat bf16

    __bf16* wt1s   = (__bf16*)(ws + 0 * WT_SZ);
    __bf16* wt1n   = (__bf16*)(ws + 1 * WT_SZ);
    __bf16* wt2s   = (__bf16*)(ws + 2 * WT_SZ);
    __bf16* wt2n   = (__bf16*)(ws + 3 * WT_SZ);
    float*  deg    = (float*)(ws + DEG_OFF);
    float*  agg    = (float*)(ws + AGG_OFF);
    __bf16* xbf    = (__bf16*)(ws + XBF_OFF);   // bf16 features; layer1 out in-place
    __bf16* meanbf = (__bf16*)(ws + MBF_OFF);

    const int ND = N_NODES * D;   // 6,400,000

    // init accumulators
    hipMemsetAsync(deg, 0, N_NODES * sizeof(float), stream);
    hipMemsetAsync(agg, 0, (size_t)ND * sizeof(float), stream);

    // weight prep (bf16, transposed) + feature cast
    prep_w_kernel<<<(D * D + 255) / 256, 256, 0, stream>>>(Ws1, wt1s);
    prep_w_kernel<<<(D * D + 255) / 256, 256, 0, stream>>>(Wn1, wt1n);
    prep_w_kernel<<<(D * D + 255) / 256, 256, 0, stream>>>(Ws2, wt2s);
    prep_w_kernel<<<(D * D + 255) / 256, 256, 0, stream>>>(Wn2, wt2n);
    cast_bf16_kernel<<<(ND + 255) / 256, 256, 0, stream>>>(in_feat, xbf, ND);

    // degrees (shared by both layers)
    deg_kernel<<<(N_EDGES + 255) / 256, 256, 0, stream>>>(dst, deg, N_EDGES);

    // ---- layer 1 ----
    scatter_kernel<<<(N_EDGES * 32 + 255) / 256, 256, 0, stream>>>(
        xbf, src, dst, agg, N_EDGES);
    mean_kernel<<<(ND + 255) / 256, 256, 0, stream>>>(agg, deg, meanbf, ND);
    // relu + bf16 output, in place over xbf
    sage_gemm_kernel<true, true><<<N_NODES / 16, 256, 0, stream>>>(
        xbf, meanbf, wt1s, wt1n, b1, (void*)xbf);

    // ---- layer 2 ----
    hipMemsetAsync(agg, 0, (size_t)ND * sizeof(float), stream);
    scatter_kernel<<<(N_EDGES * 32 + 255) / 256, 256, 0, stream>>>(
        xbf, src, dst, agg, N_EDGES);
    mean_kernel<<<(ND + 255) / 256, 256, 0, stream>>>(agg, deg, meanbf, ND);
    // final f32 output
    sage_gemm_kernel<false, false><<<N_NODES / 16, 256, 0, stream>>>(
        xbf, meanbf, wt2s, wt2n, b2, d_out);
}
